// MINN_SA_78881369359057
// MI455X (gfx1250) — compile-verified, matrix operand order checked by reading, not run on hardware
//
#include <hip/hip_runtime.h>
#include <hip/hip_bf16.h>

#define B_REP 1024
#define TCR   100
#define FEAT  15
#define AA    24
#define NINP  14
#define EPSBN 1e-5f
#define XW    40            // padded row width for im2col staging (cols 24..39 == 0)
#define XELT  (FEAT * AA)   // 360 floats per instance
#define XBUF  (FEAT * XW)   // 600 floats per staging buffer

typedef __attribute__((ext_vector_type(16))) __bf16 v16bf;
typedef __attribute__((ext_vector_type(8)))  float  v8f;
typedef __attribute__((address_space(3))) const float lds_cf;   // true LDS pointer type

// filter index -> kernel size / conv group / local filter index (padded to 16)
__constant__ int c_ks[16]  = {2,2,2,3,3,3,4,4,4,5,5,6,6,7,7,7};
__constant__ int c_grp[16] = {0,0,0,1,1,1,2,2,2,3,3,4,4,5,5,5};
__constant__ int c_loc[16] = {0,1,2,0,1,2,0,1,2,0,1,0,1,0,0,0};

struct ConvParams { const float* w[6]; const float* b[6]; };

// async global -> LDS copy, tracked by ASYNCcnt (CDNA5)
#define ASYNC_G2L_B32(ldsaddr, voff, sptr)                                  \
    asm volatile("global_load_async_to_lds_b32 %0, %1, %2"                  \
                 :: "v"(ldsaddr), "v"(voff), "s"(sptr) : "memory")

// im2col gather: KH literal 0/1 -> kk/cch/t7 constant-fold; BP is an
// address-space(3) pointer so every access is ds_load [base + imm-offset].
#define GATHER_KC(KH, KC, F0, F1, BP)                                        \
    _Pragma("unroll")                                                        \
    for (int j = 0; j < 16; ++j) {                                           \
        const int k  = (j < 8) ? ((KH) * 8 + j) : (16 + (KH) * 8 + (j - 8)); \
        const int kk = (KC) * 32 + k;                                        \
        if (kk < FEAT * 7) {                                                 \
            const int cch = kk / 7, t7 = kk % 7;                             \
            F0[j] = BP[cch * XW + t7];                                       \
            F1[j] = BP[cch * XW + t7 + 16];                                  \
        } else { F0[j] = 0.0f; F1[j] = 0.0f; }                               \
    }

__global__ void __launch_bounds__(256)
minn_conv_attn(const float* __restrict__ x, ConvParams cp,
               const float* __restrict__ fc1_w, const float* __restrict__ fc1_b,
               const float* __restrict__ waout_w,
               float* __restrict__ attw_out, float* __restrict__ pooled_out)
{
    __shared__ float  sx[8][2][FEAT][XW];   // per-wave double-buffered x staging
    __shared__ __bf16 sW[4][32][16];        // [kchunk][k][filter], zero padded
    __shared__ float  sh[TCR][16];          // fc1 hidden per instance
    __shared__ float  sscore[128];
    __shared__ float  ssort[128];
    __shared__ float  sattw[TCR];
    __shared__ float  sfc1[NINP][NINP];
    __shared__ float  sfc1b[NINP];
    __shared__ float  swa[NINP];
    __shared__ float  scb[16];
    __shared__ float  stau;

    const int tid  = threadIdx.x;
    const int lane = tid & 31;
    const int wv   = tid >> 5;
    const int rep  = blockIdx.x;

    // ---- zero staging buffers (padding cols stay 0 forever) ----
    for (int e = tid; e < 8 * 2 * XBUF; e += 256) ((float*)sx)[e] = 0.0f;

    // ---- stage padded conv weights (14 x 105 -> 16 x 128, bf16) ----
    for (int e = tid; e < 4 * 32 * 16; e += 256) {
        int kc = e >> 9, rem = e & 511, k = rem >> 4, n = rem & 15;
        int kk = kc * 32 + k;
        float v = 0.0f;
        if (n < NINP && kk < FEAT * 7) {
            int cch = kk / 7, t7 = kk % 7;
            int hk  = c_ks[n];
            if (t7 < hk) v = cp.w[c_grp[n]][(c_loc[n] * FEAT + cch) * hk + t7];
        }
        sW[kc][k][n] = (__bf16)v;
    }
    for (int e = tid; e < NINP * NINP; e += 256) sfc1[e / NINP][e % NINP] = fc1_w[e];
    if (tid < 16)   scb[tid]  = (tid < NINP) ? cp.b[c_grp[tid]][c_loc[tid]] : 0.0f;
    if (tid < NINP) { sfc1b[tid] = fc1_b[tid]; swa[tid] = waout_w[tid]; }
    __syncthreads();

    const int m  = lane & 15;   // row (A) / column (B) within tile
    const int kh = lane >> 4;   // half-wave K selector

    // ---- hoist B operands: weight matrix is invariant for whole block ----
    v16bf bv[4];
    #pragma unroll
    for (int kc = 0; kc < 4; ++kc) {
        #pragma unroll
        for (int j = 0; j < 8; ++j) {
            bv[kc][j]     = sW[kc][kh * 8 + j][m];
            bv[kc][j + 8] = sW[kc][16 + kh * 8 + j][m];
        }
    }
    // ---- hoist per-lane fc1 row / bias / mask / attention weight ----
    float wrow[NINP];
    #pragma unroll
    for (int ff = 0; ff < NINP; ++ff) wrow[ff] = sfc1[m][ff];
    const float bias = scb[m];
    const int   pmax = 24 - c_ks[m];
    const float fcb  = sfc1b[m];
    const float wav  = (lane < NINP) ? swa[lane] : 0.0f;

    // ---- async-copy addressing: 12 buffer-relative LDS offsets + 1 base ----
    unsigned ldsrel[12];
    const unsigned base0 = (unsigned)(uintptr_t)(&sx[wv][0][0][0]);
    #pragma unroll
    for (int i = 0; i < 12; ++i) {
        int e = lane + i * 32; if (e > XELT - 1) e = XELT - 1;  // clamp: dup write, same data
        int cch = e / AA, a = e - cch * AA;
        ldsrel[i] = (unsigned)((cch * XW + a) * 4);
    }
    const unsigned l4 = (unsigned)(lane * 4);

    // per-lane LDS gather bases (32-bit LDS byte offsets -> AS(3) pointers)
    const unsigned gb0 = base0 + (unsigned)(m * 4);
    lds_cf* bpA = (lds_cf*)(uintptr_t)gb0;                         // buffer 0
    lds_cf* bpB = (lds_cf*)(uintptr_t)(gb0 + (unsigned)(XBUF * 4)); // buffer 1

    const float* xrep = x + (size_t)rep * TCR * XELT;

    // ---- prologue: kick off first tile into buffer 0 ----
    if (wv < TCR) {
        const float* xi = xrep + (size_t)wv * XELT;
        #pragma unroll
        for (int i = 0; i < 12; ++i) {
            unsigned off = l4 + (unsigned)(i * 128);
            if (i == 11) off = (off > (XELT - 1) * 4u) ? (XELT - 1) * 4u : off;
            ASYNC_G2L_B32(base0 + ldsrel[i], off, xi);
        }
    }

    int buf = 0;
    for (int t = wv; t < TCR; t += 8) {
        int tn = t + 8;
        if (tn < TCR) {                       // wave-uniform: issue next tile
            asm volatile("s_wait_dscnt 0" ::: "memory");  // prior reads of dest buffer done
            const float* xi = xrep + (size_t)tn * XELT;
            const unsigned nb = base0 + (buf ? 0u : (unsigned)(XBUF * 4));
            #pragma unroll
            for (int i = 0; i < 12; ++i) {
                unsigned off = l4 + (unsigned)(i * 128);
                if (i == 11) off = (off > (XELT - 1) * 4u) ? (XELT - 1) * 4u : off;
                ASYNC_G2L_B32(nb + ldsrel[i], off, xi);
            }
            asm volatile("s_wait_asynccnt 12" ::: "memory");  // current tile landed
        } else {
            asm volatile("s_wait_asynccnt 0" ::: "memory");
        }

        lds_cf* bp = buf ? bpB : bpA;   // one live address register; offsets immediate

        v8f acc0 = {0,0,0,0,0,0,0,0};
        v8f acc1 = {0,0,0,0,0,0,0,0};
        #pragma unroll
        for (int kc = 0; kc < 4; ++kc) {
            float f0[16], f1[16];
            if (kh == 0) { GATHER_KC(0, kc, f0, f1, bp); }
            else         { GATHER_KC(1, kc, f0, f1, bp); }
            v16bf a0, a1;
            #pragma unroll
            for (int j = 0; j < 16; ++j) { a0[j] = (__bf16)f0[j]; a1[j] = (__bf16)f1[j]; }
            acc0 = __builtin_amdgcn_wmma_f32_16x16x32_bf16(false, a0, false, bv[kc],
                                                           (short)0, acc0, false, false);
            acc1 = __builtin_amdgcn_wmma_f32_16x16x32_bf16(false, a1, false, bv[kc],
                                                           (short)0, acc1, false, false);
        }

        // bias + relu + masked global max over valid positions
        float fm = -1e30f;
        #pragma unroll
        for (int r = 0; r < 8; ++r) {
            int p0 = kh * 8 + r;
            int p1 = 16 + kh * 8 + r;
            float y0 = fmaxf(acc0[r] + bias, 0.0f);
            float y1 = fmaxf(acc1[r] + bias, 0.0f);
            if (p0 <= pmax) fm = fmaxf(fm, y0);
            if (p1 <= pmax) fm = fmaxf(fm, y1);
        }
        fm = fmaxf(fm, __shfl_xor(fm, 16, 32));   // merge row halves (wave32)

        // fc1 + relu (feature f lives in lane f; broadcast via shuffles)
        float hv = fcb;
        #pragma unroll
        for (int ff = 0; ff < NINP; ++ff)
            hv += __shfl(fm, ff, 32) * wrow[ff];
        hv = fmaxf(hv, 0.0f);
        if (lane < NINP) sh[t][lane] = hv;

        float sc = hv * wav;
        #pragma unroll
        for (int msk = 16; msk > 0; msk >>= 1) sc += __shfl_xor(sc, msk, 32);
        if (lane == 0) sscore[t] = sc;

        buf ^= 1;
    }
    __syncthreads();

    // ---- sparsemax over 100 scores: bitonic sort (desc) on 128-pad ----
    if (tid < 128) ssort[tid] = (tid < TCR) ? sscore[tid] : -3.0e38f;
    __syncthreads();
    for (int k = 2; k <= 128; k <<= 1) {
        for (int j = k >> 1; j > 0; j >>= 1) {
            if (tid < 128) {
                int ixj = tid ^ j;
                if (ixj > tid) {
                    float a = ssort[tid], b = ssort[ixj];
                    bool up = ((tid & k) == 0);
                    if (up ? (a < b) : (a > b)) { ssort[tid] = b; ssort[ixj] = a; }
                }
            }
            __syncthreads();
        }
    }
    if (tid == 0) {
        float cum = 0.0f, csel = 0.0f; int ksel = 1;
        for (int j = 0; j < TCR; ++j) {
            float zj = ssort[j];
            cum += zj;
            if (1.0f + (float)(j + 1) * zj > cum) { ksel = j + 1; csel = cum; }
        }
        stau = (csel - 1.0f) / (float)ksel;
    }
    __syncthreads();
    if (tid < TCR) {
        float w = fmaxf(sscore[tid] - stau, 0.0f);
        sattw[tid] = w;
        attw_out[rep * TCR + tid] = w;
    }
    __syncthreads();
    if (tid < NINP) {
        float p = 0.0f;
        for (int t = 0; t < TCR; ++t) p += sattw[t] * sh[t][tid];
        pooled_out[rep * NINP + tid] = p;
    }
}

__global__ void __launch_bounds__(1024)
minn_decode(const float* __restrict__ pooled,
            const float* __restrict__ decf_w, const float* __restrict__ decf_b,
            const float* __restrict__ bn_g,   const float* __restrict__ bn_b,
            const float* __restrict__ decs_w, const float* __restrict__ decs_b,
            float* __restrict__ logits)
{
    __shared__ float ssum[NINP], ssq[NINP];
    __shared__ float swg[NINP * NINP], swb[NINP], sg[NINP], sb[NINP], ssw[2 * NINP], ssb[2];
    const int tid = threadIdx.x;              // tid == batch index b
    if (tid < NINP * NINP) swg[tid] = decf_w[tid];
    if (tid < NINP) { swb[tid] = decf_b[tid]; sg[tid] = bn_g[tid]; sb[tid] = bn_b[tid];
                      ssum[tid] = 0.0f; ssq[tid] = 0.0f; }
    if (tid < 2 * NINP) ssw[tid] = decs_w[tid];
    if (tid < 2)        ssb[tid] = decs_b[tid];
    __syncthreads();

    float pv[NINP], z[NINP];
    #pragma unroll
    for (int f = 0; f < NINP; ++f) pv[f] = pooled[tid * NINP + f];
    #pragma unroll
    for (int d = 0; d < NINP; ++d) {
        float acc = swb[d];
        #pragma unroll
        for (int f = 0; f < NINP; ++f) acc += pv[f] * swg[d * NINP + f];
        z[d] = acc;
    }
    // batch statistics over B=1024 via wave32 shuffle reduce + shared atomics
    #pragma unroll
    for (int d = 0; d < NINP; ++d) {
        float s = z[d], q = z[d] * z[d];
        #pragma unroll
        for (int msk = 16; msk > 0; msk >>= 1) {
            s += __shfl_xor(s, msk, 32);
            q += __shfl_xor(q, msk, 32);
        }
        if ((tid & 31) == 0) { atomicAdd(&ssum[d], s); atomicAdd(&ssq[d], q); }
    }
    __syncthreads();
    float l0 = ssb[0], l1 = ssb[1];
    #pragma unroll
    for (int d = 0; d < NINP; ++d) {
        float mu  = ssum[d] * (1.0f / B_REP);
        float var = ssq[d] * (1.0f / B_REP) - mu * mu;
        float zn  = fmaxf((z[d] - mu) * rsqrtf(var + EPSBN) * sg[d] + sb[d], 0.0f);
        l0 += zn * ssw[d];
        l1 += zn * ssw[NINP + d];
    }
    logits[tid * 2 + 0] = l0;
    logits[tid * 2 + 1] = l1;
}

extern "C" void kernel_launch(void* const* d_in, const int* in_sizes, int n_in,
                              void* d_out, int out_size, void* d_ws, size_t ws_size,
                              hipStream_t stream) {
    const float* x = (const float*)d_in[0];
    ConvParams cp;
    for (int g = 0; g < 6; ++g) {
        cp.w[g] = (const float*)d_in[1 + 2 * g];
        cp.b[g] = (const float*)d_in[2 + 2 * g];
    }
    const float* fc1_w  = (const float*)d_in[13];
    const float* fc1_b  = (const float*)d_in[14];
    const float* waout  = (const float*)d_in[15];
    const float* decf_w = (const float*)d_in[16];
    const float* decf_b = (const float*)d_in[17];
    const float* bn_g   = (const float*)d_in[18];
    const float* bn_b   = (const float*)d_in[19];
    const float* decs_w = (const float*)d_in[20];
    const float* decs_b = (const float*)d_in[21];

    float* logits = (float*)d_out;                 // (1024, 2)
    float* attw   = (float*)d_out + B_REP * 2;     // (1024, 100)
    float* pooled = (float*)d_ws;                  // (1024, 14) scratch

    minn_conv_attn<<<B_REP, 256, 0, stream>>>(x, cp, fc1_w, fc1_b, waout, attw, pooled);
    minn_decode<<<1, 1024, 0, stream>>>(pooled, decf_w, decf_b, bn_g, bn_b,
                                        decs_w, decs_b, logits);
}
